// NomadicMoE_15702400434428
// MI455X (gfx1250) — matrix operand order repaired and checked by reading, not compiled, not run on hardware
//
#include <hip/hip_runtime.h>
#include <cstdint>

// ---------------------------------------------------------------------------
// NomadicMoE forward for MI455X (gfx1250): f16 WMMA GEMM pipeline with
// TDM (tensor_load_to_lds) double-buffered tile staging.
// ---------------------------------------------------------------------------

typedef __attribute__((ext_vector_type(16))) _Float16 v16h;
typedef __attribute__((ext_vector_type(8)))  _Float16 v8h;
typedef __attribute__((ext_vector_type(8)))  float    v8f;
typedef __attribute__((ext_vector_type(4)))  unsigned int u32x4;
typedef __attribute__((ext_vector_type(8)))  int          i32x8;
typedef __attribute__((ext_vector_type(4)))  int          i32x4;

#define Bsz 8192
#define Dsz 512
#define Hsz 1024
#define Osz 512
#define Esz 16
#define Gsz 256

#if defined(__AMDGCN__) && __has_builtin(__builtin_amdgcn_tensor_load_to_lds) && \
    __has_builtin(__builtin_amdgcn_s_wait_tensorcnt)
#define USE_TDM 1
#else
#define USE_TDM 0
#endif

// ---------------------------------------------------------------------------
// f32 -> f16 elementwise convert (vectorized by 4)
// ---------------------------------------------------------------------------
struct H4 { _Float16 a, b, c, d; };

__global__ void cvt_f32_to_f16(const float* __restrict__ in,
                               _Float16* __restrict__ out, int n4) {
  int i = blockIdx.x * blockDim.x + threadIdx.x;
  if (i < n4) {
    float4 v = ((const float4*)in)[i];
    H4 h;
    h.a = (_Float16)v.x; h.b = (_Float16)v.y;
    h.c = (_Float16)v.z; h.d = (_Float16)v.w;
    ((H4*)out)[i] = h;
  }
}

// ---------------------------------------------------------------------------
// Batched tiled transpose + f32->f16: in[e][R][C] f32 -> out[e][C][R] f16
// ---------------------------------------------------------------------------
__global__ void transpose_cvt(const float* __restrict__ in,
                              _Float16* __restrict__ out, int R, int C) {
  __shared__ float tile[32][33];
  int e = blockIdx.z;
  const float* src = in + (size_t)e * R * C;
  _Float16* dst = out + (size_t)e * C * R;
  int c0 = blockIdx.x * 32, r0 = blockIdx.y * 32;
  for (int i = threadIdx.y; i < 32; i += 8) {
    int r = r0 + i, c = c0 + threadIdx.x;
    tile[i][threadIdx.x] = (r < R && c < C) ? src[(size_t)r * C + c] : 0.0f;
  }
  __syncthreads();
  for (int i = threadIdx.y; i < 32; i += 8) {
    int c = c0 + i, r = r0 + threadIdx.x;
    if (c < C && r < R) dst[(size_t)c * R + r] = (_Float16)tile[threadIdx.x][i];
  }
}

// ---------------------------------------------------------------------------
// Fragment gather from LDS: two 16-byte chunks -> v16h
// ---------------------------------------------------------------------------
__device__ __forceinline__ v16h ld_frag(const _Float16* lo_p, const _Float16* hi_p) {
  v8h lo = *(const v8h*)lo_p;
  v8h hi = *(const v8h*)hi_p;
  return __builtin_shufflevector(lo, hi, 0, 1, 2, 3, 4, 5, 6, 7,
                                         8, 9, 10, 11, 12, 13, 14, 15);
}

#if USE_TDM
// ---------------------------------------------------------------------------
// TDM 2D tile load: rows x 64 halfs from row-major f16 (row stride in halfs)
// into padded LDS rows of 72 halfs (144 B). D# per CDNA5 ISA ch.8:
//   group0: count=1 | lds_addr | global_addr(57b) | type=2
//   group1: data_size=2B, pad_enable, pad_interval=32 DWORDs, pad_amount=4
//           DWORDs, tensor_dim0/1 = remaining extents (OOB reads -> 0),
//           tile_dim0=64, tile_dim1=rows, tensor_dim0_stride = row stride.
// ---------------------------------------------------------------------------
__device__ __forceinline__ void tdm_tile_load(const _Float16* gsrc,
                                              unsigned lds_byte_off,
                                              int rows_rem, int cols_rem,
                                              int tile_rows, int stride_halfs) {
  uint64_t ga = (uint64_t)(uintptr_t)gsrc;
  u32x4 g0;
  g0[0] = 1u;                                   // count = 1 valid descriptor
  g0[1] = lds_byte_off;                         // lds_addr
  g0[2] = (unsigned)ga;                         // global_addr[31:0]
  g0[3] = ((unsigned)(ga >> 32) & 0x1FFFFFFu)   // global_addr[56:32]
        | (2u << 30);                           // type = 2 ("image")
  unsigned ucols = (unsigned)cols_rem;
  unsigned urows = (unsigned)(rows_rem < 0 ? 0 : rows_rem);
  i32x8 g1;
  g1[0] = (int)((1u << 16)                      // data_size = 2 bytes
              | (1u << 20)                      // pad_enable
              | (4u << 22)                      // pad_interval: 32 DWORDs
              | (3u << 25));                    // pad_amount: 4 DWORDs (16 B)
  g1[1] = (int)((ucols & 0xFFFFu) << 16);       // tensor_dim0[15:0] @ bits 63:48
  g1[2] = (int)((ucols >> 16) | ((urows & 0xFFFFu) << 16)); // dim0 hi / dim1 lo
  g1[3] = (int)(((urows >> 16) & 0xFFFFu) | (64u << 16));   // dim1 hi / tile_dim0=64
  g1[4] = (int)(unsigned)tile_rows;             // tile_dim1 ; tile_dim2 = 0
  g1[5] = (int)(unsigned)stride_halfs;          // tensor_dim0_stride[31:0]
  g1[6] = 0;                                    // stride hi / dim1_stride lo
  g1[7] = 0;
  i32x4 gz = {0, 0, 0, 0};
#if __clang_major__ >= 23
  i32x8 gz8 = {0, 0, 0, 0, 0, 0, 0, 0};
  __builtin_amdgcn_tensor_load_to_lds(g0, g1, gz, gz, gz8, 0);
#else
  __builtin_amdgcn_tensor_load_to_lds(g0, g1, gz, gz, 0);
#endif
}
#endif  // USE_TDM

// ---------------------------------------------------------------------------
// WMMA GEMM:  C[M,N] = act( A[M,K] * Bt[N,K]^T + bias [+ u1*v1^T + u2*v2^T] )
// A, Bt f16; accum f32; output f32 (Cf) or f16 (Ch).
// Block tile 128x64, BK=64, 256 threads = 8 waves; wave tile 32(M) x 32(N).
// TDM path: double-buffered LDS stages, DMA of tile k+1 overlaps compute of k.
// ACT: 0 = none, 1 = relu, 2 = tanh
// ---------------------------------------------------------------------------
template <int ACT, bool OUT16, bool RANK2>
__global__ __launch_bounds__(256)
void gemm_wmma(const _Float16* __restrict__ A, int lda,
               const _Float16* __restrict__ Bt, int ldb,
               const float* __restrict__ bias,
               float* __restrict__ Cf, _Float16* __restrict__ Ch, int ldc,
               int M, int N, int K,
               const float* __restrict__ u1, const float* __restrict__ v1,
               const float* __restrict__ u2, const float* __restrict__ v2) {
  constexpr int BM = 128, BN = 64, BK = 64, LDSS = BK + 8;  // 144B rows, 16B aligned
  __shared__ _Float16 As[2][BM][LDSS];
  __shared__ _Float16 Bs[2][BN][LDSS];

  const int tid  = threadIdx.x;
  const int lane = tid & 31;
  const int wave = tid >> 5;
  const int wm = wave >> 1;       // 0..3 : M sub-tile group (32 rows each)
  const int wn = wave & 1;        // 0..1 : N sub-tile group (32 cols each)
  const int bm = blockIdx.x * BM;
  const int bn = blockIdx.y * BN;

  // A fragment (ISA 16-bit A layout): lanes 0-15 rows; half-wave selects
  // K base {0,8}; two 16B chunks at +0 and +16 halfs.
  const int mrow = lane & 15;
  const int kbA  = (lane >> 4) << 3;   // 0 or 8
  // B fragment: column = lane&15; same K layout as A (base {0,8}, chunks +0/+16).
  const int kbB  = (lane >> 4) << 3;   // 0 or 8

  v8f acc[2][2] = {};
  const int nk = K / BK;

#if USE_TDM
  for (int it = 0; it < nk; ++it) {
    const int pb = it & 1;
    if (tid < 32) {                       // wave 0 drives the DMA engine
      if (it == 0) {
        tdm_tile_load(A  + (size_t)bm * lda, (unsigned)(uintptr_t)&As[0][0][0],
                      M - bm, K, BM, lda);
        tdm_tile_load(Bt + (size_t)bn * ldb, (unsigned)(uintptr_t)&Bs[0][0][0],
                      N - bn, K, BN, ldb);
      }
      const int k1 = (it + 1) * BK;
      if (k1 < K) {                       // prefetch next tile into other stage
        tdm_tile_load(A  + (size_t)bm * lda + k1,
                      (unsigned)(uintptr_t)&As[pb ^ 1][0][0], M - bm, K - k1, BM, lda);
        tdm_tile_load(Bt + (size_t)bn * ldb + k1,
                      (unsigned)(uintptr_t)&Bs[pb ^ 1][0][0], N - bn, K - k1, BN, ldb);
        __builtin_amdgcn_s_wait_tensorcnt(2);   // current tile landed
      } else {
        __builtin_amdgcn_s_wait_tensorcnt(0);
      }
    }
    __syncthreads();                       // tile pb visible to all waves
#pragma unroll
    for (int s = 0; s < 2; ++s) {          // two K=32 sub-steps
      v16h af[2], bf[2];
#pragma unroll
      for (int i = 0; i < 2; ++i) {
        const _Float16* p = &As[pb][wm * 32 + i * 16 + mrow][s * 32 + kbA];
        af[i] = ld_frag(p, p + 16);
      }
#pragma unroll
      for (int j = 0; j < 2; ++j) {
        const _Float16* q = &Bs[pb][wn * 32 + j * 16 + mrow][s * 32 + kbB];
        bf[j] = ld_frag(q, q + 16);
      }
#pragma unroll
      for (int i = 0; i < 2; ++i)
#pragma unroll
        for (int j = 0; j < 2; ++j)
          acc[i][j] = __builtin_amdgcn_wmma_f32_16x16x32_f16(
              false, af[i], false, bf[j], (short)0, acc[i][j], false, false);
    }
    __syncthreads();                       // all reads of stage pb done
  }
#else
  // Fallback: synchronous single-stage tile staging through VGPRs.
  for (int it = 0; it < nk; ++it) {
    const int k0 = it * BK;
    __syncthreads();
#pragma unroll
    for (int i = 0; i < 4; ++i) {          // A: 128x64 halfs = 1024 16B chunks
      int c = tid + i * 256;
      int r = c >> 3, col = (c & 7) << 3;
      *(uint4*)(&As[0][r][col]) =
          *(const uint4*)(A + (size_t)(bm + r) * lda + k0 + col);
    }
#pragma unroll
    for (int i = 0; i < 2; ++i) {          // B: 64x64 halfs = 512 chunks (masked)
      int c = tid + i * 256;
      int r = c >> 3, col = (c & 7) << 3;
      int n = bn + r;
      uint4 val = {0u, 0u, 0u, 0u};
      if (n < N) val = *(const uint4*)(Bt + (size_t)n * ldb + k0 + col);
      *(uint4*)(&Bs[0][r][col]) = val;
    }
    if (k0 + BK < K) {
      __builtin_prefetch(A + (size_t)(bm + (tid >> 1)) * lda + k0 + BK + (tid & 1) * 32, 0, 0);
      int pn = bn + (tid >> 2);
      if (pn < N)
        __builtin_prefetch(Bt + (size_t)pn * ldb + k0 + BK + (tid & 3) * 16, 0, 0);
    }
    __syncthreads();
#pragma unroll
    for (int s = 0; s < 2; ++s) {
      v16h af[2], bf[2];
#pragma unroll
      for (int i = 0; i < 2; ++i) {
        const _Float16* p = &As[0][wm * 32 + i * 16 + mrow][s * 32 + kbA];
        af[i] = ld_frag(p, p + 16);
      }
#pragma unroll
      for (int j = 0; j < 2; ++j) {
        const _Float16* q = &Bs[0][wn * 32 + j * 16 + mrow][s * 32 + kbB];
        bf[j] = ld_frag(q, q + 16);
      }
#pragma unroll
      for (int i = 0; i < 2; ++i)
#pragma unroll
        for (int j = 0; j < 2; ++j)
          acc[i][j] = __builtin_amdgcn_wmma_f32_16x16x32_f16(
              false, af[i], false, bf[j], (short)0, acc[i][j], false, false);
    }
  }
#endif

  // ---- epilogue: C tile layout = VGPR r -> (M = r + 8*(lane>=16), N = lane&15)
  const int moff = (lane >> 4) << 3;
  const int ncol = lane & 15;
#pragma unroll
  for (int i = 0; i < 2; ++i) {
#pragma unroll
    for (int j = 0; j < 2; ++j) {
      int gn = bn + wn * 32 + j * 16 + ncol;
      if (gn >= N) continue;
      float bv = bias[gn];
      float v1n = 0.f, v2n = 0.f;
      if constexpr (RANK2) { v1n = v1[gn]; v2n = v2[gn]; }
#pragma unroll
      for (int r = 0; r < 8; ++r) {
        int gm = bm + wm * 32 + i * 16 + moff + r;
        float v = acc[i][j][r] + bv;
        if constexpr (RANK2) v += u1[gm] * v1n + u2[gm] * v2n;
        if constexpr (ACT == 1) v = fmaxf(v, 0.0f);
        if constexpr (ACT == 2) v = tanhf(v);
        if constexpr (OUT16) Ch[(size_t)gm * ldc + gn] = (_Float16)v;
        else                 Cf[(size_t)gm * ldc + gn] = v;
      }
    }
  }
}

// ---------------------------------------------------------------------------
// Softmax over E=16 per row (integer temperature input)
// ---------------------------------------------------------------------------
__global__ void softmax16_kernel(const float* __restrict__ logits,
                                 float* __restrict__ probs,
                                 const int* __restrict__ tptr, int B) {
  int b = blockIdx.x * blockDim.x + threadIdx.x;
  if (b >= B) return;
  float invT = 1.0f / (float)(*tptr);
  float v[Esz];
  float mx = -3.4e38f;
#pragma unroll
  for (int e = 0; e < Esz; ++e) {
    v[e] = logits[(size_t)b * Esz + e] * invT;
    mx = fmaxf(mx, v[e]);
  }
  float s = 0.0f;
#pragma unroll
  for (int e = 0; e < Esz; ++e) { v[e] = expf(v[e] - mx); s += v[e]; }
  float r = 1.0f / s;
#pragma unroll
  for (int e = 0; e < Esz; ++e) probs[(size_t)b * Esz + e] = v[e] * r;
}

// ---------------------------------------------------------------------------
// y[b,o] = sum_e probs[b,e] * eo[b,e,o]   (bandwidth-bound)
// ---------------------------------------------------------------------------
__global__ void combine_kernel(const float* __restrict__ probs,
                               const float* __restrict__ eo,
                               float* __restrict__ y, int O) {
  int o = blockIdx.x * blockDim.x + threadIdx.x;
  int b = blockIdx.y;
  const float* row = eo + (size_t)b * Esz * O + o;
  float s = 0.0f;
#pragma unroll
  for (int e = 0; e < Esz; ++e) s += probs[(size_t)b * Esz + e] * row[(size_t)e * O];
  y[(size_t)b * O + o] = s;
}

// ---------------------------------------------------------------------------
// Host launcher
// ---------------------------------------------------------------------------
extern "C" void kernel_launch(void* const* d_in, const int* in_sizes, int n_in,
                              void* d_out, int out_size, void* d_ws, size_t ws_size,
                              hipStream_t stream) {
  (void)in_sizes; (void)n_in; (void)out_size; (void)ws_size;
  const float* x   = (const float*)d_in[0];
  const float* dh  = (const float*)d_in[1];
  const float* de  = (const float*)d_in[2];
  const int*   tp  = (const int*)d_in[3];
  const float* W1  = (const float*)d_in[4];
  const float* b1  = (const float*)d_in[5];
  const float* W2  = (const float*)d_in[6];
  const float* b2  = (const float*)d_in[7];
  const float* W3  = (const float*)d_in[8];
  const float* b3  = (const float*)d_in[9];
  const float* Wg1 = (const float*)d_in[10];
  const float* bg1 = (const float*)d_in[11];
  const float* Wg2 = (const float*)d_in[12];
  const float* bg2 = (const float*)d_in[13];
  const float* Wg3 = (const float*)d_in[14];
  const float* bg3 = (const float*)d_in[15];

  // d_out layout: y_hat | gate_probs | gate_logits | expert_outputs
  float* y_hat  = (float*)d_out;
  float* probs  = y_hat + (size_t)Bsz * Osz;
  float* logits = probs + (size_t)Bsz * Esz;
  float* eo     = logits + (size_t)Bsz * Esz;

  // workspace carve-up (256B aligned), ~118 MB total
  char* wp = (char*)d_ws;
  auto carve = [&](size_t bytes) -> char* {
    char* r = wp; wp += (bytes + 255) & ~(size_t)255; return r;
  };
  _Float16* x16  = (_Float16*)carve((size_t)Bsz * Dsz * 2);
  _Float16* w1t  = (_Float16*)carve((size_t)Esz * Hsz * Dsz * 2);
  _Float16* w2t  = (_Float16*)carve((size_t)Esz * Hsz * Hsz * 2);
  _Float16* w3t  = (_Float16*)carve((size_t)Esz * Osz * Hsz * 2);
  _Float16* wg1t = (_Float16*)carve((size_t)Gsz * 512 * 2);
  _Float16* wg2t = (_Float16*)carve((size_t)Gsz * Gsz * 2);
  _Float16* wg3t = (_Float16*)carve((size_t)Esz * Gsz * 2);
  _Float16* g1   = (_Float16*)carve((size_t)Bsz * Gsz * 2);
  _Float16* g2   = (_Float16*)carve((size_t)Bsz * Gsz * 2);
  _Float16* h1   = (_Float16*)carve((size_t)Bsz * Hsz * 2);
  _Float16* h2   = (_Float16*)carve((size_t)Bsz * Hsz * 2);

  dim3 tb(32, 8);
  // convert activations + all weights to f16 (weights transposed to [N][K];
  // all 16 experts' f16 weights total 64 MB -> resident in the 192 MB L2)
  cvt_f32_to_f16<<<(Bsz * Dsz / 4 + 255) / 256, 256, 0, stream>>>(x, x16, Bsz * Dsz / 4);
  transpose_cvt<<<dim3(Hsz / 32, Dsz / 32, Esz), tb, 0, stream>>>(W1, w1t, Dsz, Hsz);
  transpose_cvt<<<dim3(Hsz / 32, Hsz / 32, Esz), tb, 0, stream>>>(W2, w2t, Hsz, Hsz);
  transpose_cvt<<<dim3(Osz / 32, Hsz / 32, Esz), tb, 0, stream>>>(W3, w3t, Hsz, Osz);
  transpose_cvt<<<dim3(Gsz / 32, 512 / 32, 1), tb, 0, stream>>>(Wg1, wg1t, 512, Gsz);
  transpose_cvt<<<dim3(Gsz / 32, Gsz / 32, 1), tb, 0, stream>>>(Wg2, wg2t, Gsz, Gsz);
  transpose_cvt<<<dim3(1, Gsz / 32, 1), tb, 0, stream>>>(Wg3, wg3t, Gsz, Esz);

  dim3 blk(256);
  // ---- gate MLP; delta_hybrid/delta_err columns folded in as a rank-2
  //      epilogue correction on layer 1.
  gemm_wmma<1, true, true><<<dim3(Bsz / 128, Gsz / 64), blk, 0, stream>>>(
      x16, Dsz, wg1t, 512, bg1, nullptr, g1, Gsz, Bsz, Gsz, 512,
      dh, Wg1 + (size_t)512 * Gsz, de, Wg1 + (size_t)513 * Gsz);
  gemm_wmma<1, true, false><<<dim3(Bsz / 128, Gsz / 64), blk, 0, stream>>>(
      g1, Gsz, wg2t, Gsz, bg2, nullptr, g2, Gsz, Bsz, Gsz, Gsz,
      nullptr, nullptr, nullptr, nullptr);
  gemm_wmma<0, false, false><<<dim3(Bsz / 128, 1), blk, 0, stream>>>(
      g2, Gsz, wg3t, Gsz, bg3, logits, nullptr, Esz, Bsz, Esz, Gsz,
      nullptr, nullptr, nullptr, nullptr);
  softmax16_kernel<<<(Bsz + 255) / 256, 256, 0, stream>>>(logits, probs, tp, Bsz);

  // ---- all-expert dense MLP
  for (int e = 0; e < Esz; ++e) {
    gemm_wmma<2, true, false><<<dim3(Bsz / 128, Hsz / 64), blk, 0, stream>>>(
        x16, Dsz, w1t + (size_t)e * Hsz * Dsz, Dsz, b1 + (size_t)e * Hsz,
        nullptr, h1, Hsz, Bsz, Hsz, Dsz, nullptr, nullptr, nullptr, nullptr);
    gemm_wmma<2, true, false><<<dim3(Bsz / 128, Hsz / 64), blk, 0, stream>>>(
        h1, Hsz, w2t + (size_t)e * Hsz * Hsz, Hsz, b2 + (size_t)e * Hsz,
        nullptr, h2, Hsz, Bsz, Hsz, Hsz, nullptr, nullptr, nullptr, nullptr);
    gemm_wmma<0, false, false><<<dim3(Bsz / 128, Osz / 64), blk, 0, stream>>>(
        h2, Hsz, w3t + (size_t)e * Osz * Hsz, Hsz, b3 + (size_t)e * Osz,
        eo + (size_t)e * Osz, nullptr, Esz * Osz, Bsz, Osz, Hsz,
        nullptr, nullptr, nullptr, nullptr);
  }

  // ---- soft routing combine
  combine_kernel<<<dim3(Osz / 256, Bsz), 256, 0, stream>>>(probs, eo, y_hat, Osz);
}
